// GraphBlock_26749056319740
// MI455X (gfx1250) — compile-verified, matrix operand order checked by reading, not run on hardware
//
#include <hip/hip_runtime.h>
#include <hip/hip_bf16.h>

#define NNODES 8192
#define FDIM   128

typedef __attribute__((ext_vector_type(16))) __bf16 v16bf;
typedef __attribute__((ext_vector_type(8)))  float  v8f;
typedef unsigned int uint32x4 __attribute__((ext_vector_type(4)));
typedef int          int32x8  __attribute__((ext_vector_type(8)));
typedef int          int32x4  __attribute__((ext_vector_type(4)));

// ---------------------------------------------------------------------------
// WMMA fragment helpers (wave32, v_wmma_f32_16x16x32_bf16)
// A-frag 16x32: lane l -> row m=l&15; holds K = kb..kb+7 and 16+kb..16+kb+7,
//               kb = (l>=16)*8.  (ISA 7.12.2 "16-bit A-Matrix 16x32")
// B-frag 32x16: mirrored, lane l -> col n=l&15, same K split.
// C/D  16x16 : lane l -> col n=l&15, VGPR r -> row (l>=16?8:0)+r.
// ---------------------------------------------------------------------------
__device__ __forceinline__ v16bf load_A_frag(const float* __restrict__ src, int lda, int lane) {
  int m = lane & 15, kb = (lane >> 4) << 3;
  const float* p = src + (size_t)m * lda;
  v16bf a;
#pragma unroll
  for (int i = 0; i < 8; ++i) a[i] = (__bf16)p[kb + i];
#pragma unroll
  for (int i = 0; i < 8; ++i) a[8 + i] = (__bf16)p[16 + kb + i];
  return a;
}

__device__ __forceinline__ v16bf load_B_frag(const float* __restrict__ src, int ldb, int lane) {
  int n = lane & 15, kb = (lane >> 4) << 3;
  v16bf b;
#pragma unroll
  for (int i = 0; i < 8; ++i) b[i] = (__bf16)src[(size_t)(kb + i) * ldb + n];
#pragma unroll
  for (int i = 0; i < 8; ++i) b[8 + i] = (__bf16)src[(size_t)(16 + kb + i) * ldb + n];
  return b;
}

// B fragment gathered from an LDS-resident row-major bf16 tile [32 x FDIM]
__device__ __forceinline__ v16bf lds_B_frag(const __bf16* tile, int ct, int lane) {
  int n = lane & 15, kb = (lane >> 4) << 3;
  const __bf16* p = tile + ct * 16 + n;
  v16bf b;
#pragma unroll
  for (int i = 0; i < 8; ++i) b[i] = p[(kb + i) * FDIM];
#pragma unroll
  for (int i = 0; i < 8; ++i) b[8 + i] = p[(16 + kb + i) * FDIM];
  return b;
}

// ---------------------------------------------------------------------------
// TDM: DMA a [rows x 128] bf16 tile (row-major, stride 128) into LDS.
// D# built per ISA 8.3/8.4: group0 {count=1, lds_addr, global_addr, type=2},
// group1 {data_size=2B, tensor_dim0=128, tensor_dim1=8192, tile=128 x rows,
// dim0_stride=128}. Issued by one wave; completion via s_wait_tensorcnt.
// This toolchain exposes the 6-arg builtin:
//   (uint32x4 g0, int32x8 g1, int32x4, int32x4, int32x8, i32 cpol)
// ---------------------------------------------------------------------------
__device__ __forceinline__ void tdm_load_tile_bf16(const __bf16* gsrc, unsigned lds_off, int rows) {
#if __has_builtin(__builtin_amdgcn_tensor_load_to_lds) && __has_builtin(__builtin_amdgcn_s_wait_tensorcnt)
  unsigned long long ga = (unsigned long long)(uintptr_t)gsrc;
  uint32x4 g0;
  g0[0] = 1u;                                                // count=1 (valid), user mode
  g0[1] = lds_off;                                           // lds_addr (bytes)
  g0[2] = (unsigned)(ga & 0xffffffffu);                      // global_addr[31:0]
  g0[3] = (unsigned)((ga >> 32) & 0x01ffffffu) | 0x80000000u;// addr[56:32] | type=2
  int32x8 g1;
  g1[0] = 0x00010000;                                        // data_size=1 (2 bytes)
  g1[1] = (FDIM & 0xffff) << 16;                             // tensor_dim0[15:0] @ bit48
  g1[2] = (NNODES & 0xffff) << 16;                           // tensor_dim1[15:0] @ bit80
  g1[3] = (NNODES >> 16) | ((FDIM & 0xffff) << 16);          // dim1[31:16] | tile_dim0 @ bit112
  g1[4] = rows & 0xffff;                                     // tile_dim1 (tile_dim2=0)
  g1[5] = FDIM;                                              // tensor_dim0_stride[31:0]
  g1[6] = 0;                                                 // stride hi / dim1_stride lo
  g1[7] = 0;
  int32x4 z4; z4[0] = 0; z4[1] = 0; z4[2] = 0; z4[3] = 0;    // groups 2/3 unused (2-D tile)
  int32x8 z8;
#pragma unroll
  for (int i = 0; i < 8; ++i) z8[i] = 0;
  __builtin_amdgcn_tensor_load_to_lds(g0, g1, z4, z4, z8, 0);
  __builtin_amdgcn_s_wait_tensorcnt(0);
#else
  (void)gsrc; (void)lds_off; (void)rows;
#endif
}

// ---------------------------------------------------------------------------
// Generic GEMM: C[.,colOff+.] = act(A[MxK] @ B[KxN] + bias), one 16x16 tile
// per wave, K stepped by 32.  f32 in / bf16 WMMA / f32 accum out.
// ---------------------------------------------------------------------------
__global__ void wmma_gemm_kernel(const float* __restrict__ A, const float* __restrict__ B,
                                 const float* __restrict__ bias, float* __restrict__ C,
                                 int M, int Nc, int K, int lda, int ldb, int ldc,
                                 int colOff, int act) {
  int lane = threadIdx.x;
  int tile = blockIdx.x * blockDim.y + threadIdx.y;
  int tilesN = Nc >> 4;
  int tm = tile / tilesN, tn = tile % tilesN;
  int row0 = tm << 4, col0 = tn << 4;
  if (row0 >= M) return;

  v8f c = {};
  for (int k0 = 0; k0 < K; k0 += 32) {
    v16bf a = load_A_frag(A + (size_t)row0 * lda + k0, lda, lane);
    v16bf b = load_B_frag(B + (size_t)k0 * ldb + col0, ldb, lane);
    c = __builtin_amdgcn_wmma_f32_16x16x32_bf16(false, a, false, b, (short)0, c, false, false);
  }
  int n = lane & 15, rb = (lane >> 4) << 3;
  float bv = bias ? bias[col0 + n] : 0.0f;
#pragma unroll
  for (int r = 0; r < 8; ++r) {
    float v = c[r] + bv;
    if (act == 1) v = v > 0.0f ? v : 0.0f;          // relu
    C[(size_t)(row0 + rb + r) * ldc + colOff + col0 + n] = v;
  }
}

// ---------------------------------------------------------------------------
// f32 -> bf16 row-major copy (V operand preconversion, done once per matrix)
// ---------------------------------------------------------------------------
__global__ void tobf16_kernel(const float* __restrict__ s, __bf16* __restrict__ d, int n) {
  int i = (blockIdx.x * blockDim.x + threadIdx.x) * 2;
  if (i < n) {
    d[i]     = (__bf16)s[i];
    d[i + 1] = (__bf16)s[i + 1];
  }
}

// ---------------------------------------------------------------------------
// f1[i] = Wh[i,:]·a[0:128], f2[i] = Wh[i,:]·a[128:256]   (wave per row)
// ---------------------------------------------------------------------------
__global__ void f12_kernel(const float* __restrict__ Wh, const float* __restrict__ a,
                           float* __restrict__ f1, float* __restrict__ f2, int N_) {
  int row = blockIdx.x * blockDim.y + threadIdx.y;
  if (row >= N_) return;
  int lane = threadIdx.x;
  const float* w = Wh + (size_t)row * FDIM;
  float s1 = 0.0f, s2 = 0.0f;
#pragma unroll
  for (int i = lane; i < FDIM; i += 32) {
    float wv = w[i];
    s1 += wv * a[i];
    s2 += wv * a[FDIM + i];
  }
#pragma unroll
  for (int off = 16; off > 0; off >>= 1) {
    s1 += __shfl_down(s1, off, 32);
    s2 += __shfl_down(s2, off, 32);
  }
  if (lane == 0) { f1[row] = s1; f2[row] = s2; }
}

// ---------------------------------------------------------------------------
// Fused GAT attention, flash-style online softmax (att matrix never in HBM):
//   out[i,:] = act( softmax_j( mask(adj, lrelu(f1_i+f2_j)) ) @ V )
// 4 waves/block share one TDM-staged 32x128 bf16 V tile in LDS per j-step.
// P tile built directly in the WMMA A-fragment register layout.
// ---------------------------------------------------------------------------
__global__ void gat_attention_kernel(const int* __restrict__ adj,
                                     const float* __restrict__ f1,
                                     const float* __restrict__ f2,
                                     const __bf16* __restrict__ Vb,
                                     int N_,
                                     float* __restrict__ out, int ldo, int colOff,
                                     int do_elu,
                                     float* __restrict__ out2, int ldo2) {
  __shared__ __bf16 vtile[32 * FDIM];   // 8 KB staged V tile
  unsigned vtile_off = (unsigned)(uintptr_t)&vtile[0];   // flat low 32 bits == LDS offset

  int lane = threadIdx.x;
  int strip = blockIdx.x * blockDim.y + threadIdx.y;
  int i0 = strip << 4;

  int m  = lane & 15;            // P row owned by this lane (A-frag layout)
  int kb = (lane >> 4) << 3;     // K sub-offset / also C-layout row base

  int row = i0 + m;
  float f1v = f1[row];
  const int* adjrow = adj + (size_t)row * N_;

  v8f acc[8];
#pragma unroll
  for (int ct = 0; ct < 8; ++ct) acc[ct] = (v8f){};

  float mrun = -3.0e38f;
  float srun = 0.0f;

  for (int j0 = 0; j0 < N_; j0 += 32) {
    __syncthreads();             // previous tile fully consumed
#if __has_builtin(__builtin_amdgcn_tensor_load_to_lds) && __has_builtin(__builtin_amdgcn_s_wait_tensorcnt)
    if (threadIdx.y == 0) {
      tdm_load_tile_bf16(Vb + (size_t)j0 * FDIM, vtile_off, 32);
    }
#else
    {   // cooperative fallback: 128 threads x 64B
      int t = threadIdx.y * 32 + lane;
      const int4* s = (const int4*)(Vb + (size_t)j0 * FDIM);
      int4* d = (int4*)vtile;
      d[t]       = s[t];
      d[t + 128] = s[t + 128];
      d[t + 256] = s[t + 256];
      d[t + 384] = s[t + 384];
    }
#endif
    __syncthreads();             // tile visible to all 4 waves

    float e[16];
    float tmax = -3.0e38f;
#pragma unroll
    for (int g = 0; g < 2; ++g) {
      int jb = j0 + g * 16 + kb;
#pragma unroll
      for (int i = 0; i < 8; ++i) {
        int j = jb + i;
        float ev = f1v + f2[j];
        ev = ev > 0.0f ? ev : 0.2f * ev;                    // leaky_relu(0.2)
        ev = (adjrow[j] > 0) ? ev : -9.0e15f;               // mask
        e[g * 8 + i] = ev;
        tmax = fmaxf(tmax, ev);
      }
    }
    // combine row max across the two lane halves that share row m
    tmax = fmaxf(tmax, __shfl_xor(tmax, 16, 32));
    float mnew  = fmaxf(mrun, tmax);
    float scale = __expf(mrun - mnew);
    mrun = mnew;
    srun *= scale;

    v16bf pfrag;
    float psum = 0.0f;
#pragma unroll
    for (int i = 0; i < 16; ++i) {
      float p = __expf(e[i] - mnew);
      psum += p;
      pfrag[i] = (__bf16)p;
    }
    srun += psum;

    // rescale accumulators (C layout rows kb..kb+7 need those rows' scales)
#pragma unroll
    for (int r = 0; r < 8; ++r) {
      float sc = __shfl(scale, kb + r, 32);
#pragma unroll
      for (int ct = 0; ct < 8; ++ct) acc[ct][r] *= sc;
    }

    // acc += P(16x32) @ V(32x128) as 8 column tiles, V from LDS (bf16)
#pragma unroll
    for (int ct = 0; ct < 8; ++ct) {
      v16bf b = lds_B_frag(vtile, ct, lane);
      acc[ct] = __builtin_amdgcn_wmma_f32_16x16x32_bf16(false, pfrag, false, b,
                                                        (short)0, acc[ct], false, false);
    }
  }

  srun += __shfl_xor(srun, 16, 32);
  float inv = 1.0f / srun;       // lane holds 1/Z for its P-row m

#pragma unroll
  for (int r = 0; r < 8; ++r) {
    float iv = __shfl(inv, kb + r, 32);
    int orow = i0 + kb + r;
#pragma unroll
    for (int ct = 0; ct < 8; ++ct) {
      float v = acc[ct][r] * iv;
      if (do_elu) v = v > 0.0f ? v : (__expf(v) - 1.0f);    // elu
      int ocol = ct * 16 + m;
      out[(size_t)orow * ldo + colOff + ocol] = v;
      if (out2) out2[(size_t)orow * ldo2 + ocol] = v;
    }
  }
}

// ---------------------------------------------------------------------------
// copy x (N x 128) into left half of combined (N x 256)
// ---------------------------------------------------------------------------
__global__ void copy_x_kernel(const float* __restrict__ x, float* __restrict__ comb, int total) {
  int idx = blockIdx.x * blockDim.x + threadIdx.x;
  if (idx >= total) return;
  int r = idx >> 7, c = idx & 127;
  comb[(size_t)r * 256 + c] = x[idx];
}

// ---------------------------------------------------------------------------
// column mean of h2 (N x 128) -> graph_emb[128]
// ---------------------------------------------------------------------------
__global__ void colmean_kernel(const float* __restrict__ h2, float* __restrict__ o, int N_) {
  __shared__ float red[256];
  int c = blockIdx.x;
  float s = 0.0f;
  for (int r = threadIdx.x; r < N_; r += blockDim.x)
    s += h2[(size_t)r * FDIM + c];
  red[threadIdx.x] = s;
  __syncthreads();
  for (int off = blockDim.x >> 1; off > 0; off >>= 1) {
    if (threadIdx.x < (unsigned)off) red[threadIdx.x] += red[threadIdx.x + off];
    __syncthreads();
  }
  if (threadIdx.x == 0) o[c] = red[0] / (float)N_;
}

// ---------------------------------------------------------------------------
extern "C" void kernel_launch(void* const* d_in, const int* in_sizes, int n_in,
                              void* d_out, int out_size, void* d_ws, size_t ws_size,
                              hipStream_t stream) {
  const int*   adj   = (const int*)  d_in[0];   // [N,N]
  const float* x     = (const float*)d_in[1];   // [N,128]
  const float* Whead = (const float*)d_in[2];   // [2,128,128]
  const float* ahead = (const float*)d_in[3];   // [2,256,1]
  const float* Wout  = (const float*)d_in[4];   // [256,128]
  const float* aout  = (const float*)d_in[5];   // [256,1]
  const float* gw1   = (const float*)d_in[6];   // [256,256]
  const float* gb1   = (const float*)d_in[7];   // [256]
  const float* gw2   = (const float*)d_in[8];   // [256,128]
  const float* gb2   = (const float*)d_in[9];   // [128]
  float* out = (float*)d_out;                   // node_emb[N*128] ++ graph_emb[128]

  const int N = NNODES;
  float* ws = (float*)d_ws;
  size_t off = 0;
  auto alloc = [&](size_t n) { float* p = ws + off; off += n; return p; };
  float* Wh0  = alloc((size_t)N * FDIM);
  float* Wh1  = alloc((size_t)N * FDIM);
  float* f1h0 = alloc(N); float* f2h0 = alloc(N);
  float* f1h1 = alloc(N); float* f2h1 = alloc(N);
  float* hcat = alloc((size_t)N * 256);
  float* Wh2  = alloc((size_t)N * FDIM);
  float* f1o  = alloc(N); float* f2o = alloc(N);
  float* comb = alloc((size_t)N * 256);
  float* h1   = alloc((size_t)N * 256);
  float* h2b  = alloc((size_t)N * FDIM);
  __bf16* Vb0 = (__bf16*)alloc((size_t)N * FDIM / 2);
  __bf16* Vb1 = (__bf16*)alloc((size_t)N * FDIM / 2);
  __bf16* Vb2 = (__bf16*)alloc((size_t)N * FDIM / 2);
  (void)ws_size; (void)n_in; (void)in_sizes; (void)out_size;

  dim3 wblk(32, 4);
  dim3 fblk(32, 8);
  int cvtGrid = (N * FDIM / 2) / 256;

  // 1) Wh per head: [8192,128] = x @ W_heads[h]; bf16 copies for attention V
  {
    int tiles = (N / 16) * (FDIM / 16);
    wmma_gemm_kernel<<<tiles / 4, wblk, 0, stream>>>(x, Whead, nullptr, Wh0,
        N, FDIM, FDIM, FDIM, FDIM, FDIM, 0, 0);
    wmma_gemm_kernel<<<tiles / 4, wblk, 0, stream>>>(x, Whead + FDIM * FDIM, nullptr, Wh1,
        N, FDIM, FDIM, FDIM, FDIM, FDIM, 0, 0);
    tobf16_kernel<<<cvtGrid, 256, 0, stream>>>(Wh0, Vb0, N * FDIM);
    tobf16_kernel<<<cvtGrid, 256, 0, stream>>>(Wh1, Vb1, N * FDIM);
  }

  // 2) attention score vectors
  f12_kernel<<<N / 8, fblk, 0, stream>>>(Wh0, ahead,        f1h0, f2h0, N);
  f12_kernel<<<N / 8, fblk, 0, stream>>>(Wh1, ahead + 256,  f1h1, f2h1, N);

  // 3) fused attention per head -> hcat (elu, concat)
  gat_attention_kernel<<<N / 64, wblk, 0, stream>>>(adj, f1h0, f2h0, Vb0, N,
      hcat, 256, 0,   1, nullptr, 0);
  gat_attention_kernel<<<N / 64, wblk, 0, stream>>>(adj, f1h1, f2h1, Vb1, N,
      hcat, 256, 128, 1, nullptr, 0);

  // 4) Wh2 = hcat @ W_out  ([8192,256]@[256,128]); bf16 copy
  {
    int tiles = (N / 16) * (FDIM / 16);
    wmma_gemm_kernel<<<tiles / 4, wblk, 0, stream>>>(hcat, Wout, nullptr, Wh2,
        N, FDIM, 256, 256, FDIM, FDIM, 0, 0);
    tobf16_kernel<<<cvtGrid, 256, 0, stream>>>(Wh2, Vb2, N * FDIM);
  }
  f12_kernel<<<N / 8, fblk, 0, stream>>>(Wh2, aout, f1o, f2o, N);

  // 5) output attention -> node_emb (elu) into d_out AND right half of combined
  gat_attention_kernel<<<N / 64, wblk, 0, stream>>>(adj, f1o, f2o, Vb2, N,
      out, FDIM, 0, 1, comb + FDIM, 256);

  // 6) combined left half = x
  copy_x_kernel<<<(N * FDIM) / 256, 256, 0, stream>>>(x, comb, N * FDIM);

  // 7) readout MLP
  {
    int tiles1 = (N / 16) * (256 / 16);
    wmma_gemm_kernel<<<tiles1 / 4, wblk, 0, stream>>>(comb, gw1, gb1, h1,
        N, 256, 256, 256, 256, 256, 0, 1);
    int tiles2 = (N / 16) * (FDIM / 16);
    wmma_gemm_kernel<<<tiles2 / 4, wblk, 0, stream>>>(h1, gw2, gb2, h2b,
        N, FDIM, 256, 256, FDIM, FDIM, 0, 1);
  }

  // 8) graph_emb = mean over rows
  colmean_kernel<<<FDIM, 256, 0, stream>>>(h2b, out + (size_t)N * FDIM, N);
}
